// ClockModel_28741921145140
// MI455X (gfx1250) — matrix operand order repaired and checked by reading, not compiled, not run on hardware
//
#include <hip/hip_runtime.h>
#include <cstdint>
#include <cstddef>

// ---------------- model constants ----------------
#define B_    16
#define IMG_  448
#define P_    16
#define HP_   28
#define NTOK_ 784
#define CB_   768
#define D_    1024
#define NH_   8
#define DH_   128
#define DFF_  4096
#define T_    64
#define R_    32
#define NPTS_ 8
#define L_    2
#define QN_   2
#define ROWS_ (B_*QN_)     /* 32 */
#define MTOK_ (B_*NTOK_)   /* 12544 */
#define MPOL_ (B_*T_*R_)   /* 32768 */
#define EPSC_ 1e-5f

typedef __attribute__((ext_vector_type(16))) __bf16        bf16x16;
typedef __attribute__((ext_vector_type(8)))  float         floatx8;
typedef __attribute__((ext_vector_type(8)))  unsigned int  uintx8;

__device__ __forceinline__ unsigned short f2bf(float f) {
  unsigned int u = __float_as_uint(f);
  u += 0x7fffu + ((u >> 16) & 1u);           // round-to-nearest-even
  return (unsigned short)(u >> 16);
}
__device__ __forceinline__ float bf2f(unsigned short h) {
  return __uint_as_float(((unsigned int)h) << 16);
}
__device__ __forceinline__ float gelu_f(float x) {
  return 0.5f * x * (1.0f + erff(x * 0.7071067811865476f));
}
__device__ __forceinline__ int clampi(int v, int lo, int hi) {
  return v < lo ? lo : (v > hi ? hi : v);
}
__device__ __forceinline__ unsigned lds_addr_u32(const void* p) {
  return (unsigned)(unsigned long long)
      (__attribute__((address_space(3))) const char*)p;
}

// ---------------- WMMA GEMM: C[MxN] = A[MxK](bf16) * W[NxK](bf16)^T + bias ----------------
// Block tile 128x128, 8 waves (4x2), each wave 32x64 = 2x4 WMMA tiles -> 8 wmma / K-step.
// Double-buffered LDS: tiles staged with CDNA5 async global->LDS b128 copies (ASYNCcnt),
// issued for tile k+1 while tile k is consumed by WMMA; one barrier per K-step.
// A falls back to guarded sync loads only when the M-tile is partial (decoder, M=32).
// All call sites use N % 128 == 0 and K % 32 == 0.
#define GM 128
#define GN 128
#define GK 32
#define LDT 40   // padded LDS row (80 B): keeps 16-B chunk alignment, non-pow2 bank stride

__global__ __launch_bounds__(256) void gemm_bf16_wmma_k(
    const unsigned short* __restrict__ A, const unsigned short* __restrict__ Bw,
    const float* __restrict__ bias, float* __restrict__ Cf,
    unsigned short* __restrict__ Cb, int M, int N, int K, int act_gelu)
{
  __shared__ __align__(16) unsigned short As[2][GM][LDT];
  __shared__ __align__(16) unsigned short Bs[2][GN][LDT];
  const int tid  = threadIdx.x;
  const int bm   = blockIdx.y * GM;
  const int bn   = blockIdx.x * GN;
  const int wave = tid >> 5, lane = tid & 31;
  const int wm   = wave >> 1, wn = wave & 1;       // 4x2 wave grid
  const int half = lane >> 4, l16 = lane & 15;
  const bool mfull = (bm + GM <= M);
  const int  nk = K / GK;

  // per-thread chunk coordinates (512 x 16B chunks per tile, 2 per thread)
  const int r0 = tid >> 2,            sub0 = tid & 3;
  const int r1 = (tid + 256) >> 2,    sub1 = (tid + 256) & 3;

  auto stage = [&](int buf, int k0) {
    // B tile: always async (rows in range: N % 128 == 0)
    {
      unsigned l0 = lds_addr_u32(&Bs[buf][r0][sub0 * 8]);
      const unsigned short* s0 = Bw + (size_t)(bn + r0) * K + k0 + sub0 * 8;
      asm volatile("global_load_async_to_lds_b128 %0, %1, off" :: "v"(l0), "v"(s0) : "memory");
      unsigned l1 = lds_addr_u32(&Bs[buf][r1][sub1 * 8]);
      const unsigned short* s1 = Bw + (size_t)(bn + r1) * K + k0 + sub1 * 8;
      asm volatile("global_load_async_to_lds_b128 %0, %1, off" :: "v"(l1), "v"(s1) : "memory");
    }
    if (mfull) {
      // A tile: async as well (uniform fast path, no guards)
      unsigned l0 = lds_addr_u32(&As[buf][r0][sub0 * 8]);
      const unsigned short* s0 = A + (size_t)(bm + r0) * K + k0 + sub0 * 8;
      asm volatile("global_load_async_to_lds_b128 %0, %1, off" :: "v"(l0), "v"(s0) : "memory");
      unsigned l1 = lds_addr_u32(&As[buf][r1][sub1 * 8]);
      const unsigned short* s1 = A + (size_t)(bm + r1) * K + k0 + sub1 * 8;
      asm volatile("global_load_async_to_lds_b128 %0, %1, off" :: "v"(l1), "v"(s1) : "memory");
    } else {
      // partial M-tile: guarded sync loads with zero fill
      uint4 v0 = {0u, 0u, 0u, 0u}, v1 = {0u, 0u, 0u, 0u};
      if (bm + r0 < M) v0 = *(const uint4*)(A + (size_t)(bm + r0) * K + k0 + sub0 * 8);
      if (bm + r1 < M) v1 = *(const uint4*)(A + (size_t)(bm + r1) * K + k0 + sub1 * 8);
      *(uint4*)(&As[buf][r0][sub0 * 8]) = v0;
      *(uint4*)(&As[buf][r1][sub1 * 8]) = v1;
    }
  };

  floatx8 acc[2][4] = {};

  stage(0, 0);
  asm volatile("s_wait_asynccnt 0x0" ::: "memory");
  __syncthreads();

  for (int kt = 0; kt < nk; ++kt) {
    const int cur = kt & 1;
    if (kt + 1 < nk) stage(cur ^ 1, (kt + 1) * GK);   // overlap copies with WMMA

    // ---- fragments per CDNA5 VGPR layouts (ISA 7.12.2), b128 LDS reads ----
    bf16x16 af[2], bfr[4];
#pragma unroll
    for (int ti = 0; ti < 2; ++ti) {
      int row = wm * 32 + ti * 16 + l16;           // A: lane = M row
      const uint4 a0 = *(const uint4*)(&As[cur][row][half * 8]);
      const uint4 a1 = *(const uint4*)(&As[cur][row][16 + half * 8]);
      uintx8 w;
      w[0] = a0.x; w[1] = a0.y; w[2] = a0.z; w[3] = a0.w;
      w[4] = a1.x; w[5] = a1.y; w[6] = a1.z; w[7] = a1.w;
      af[ti] = __builtin_bit_cast(bf16x16, w);
    }
#pragma unroll
    for (int tj = 0; tj < 4; ++tj) {
      int col = wn * 64 + tj * 16 + l16;           // B: lane = N col, elem j<8 -> K=8*half+j, j>=8 -> K=16+8*half+(j-8)
      const uint4 b0 = *(const uint4*)(&Bs[cur][col][half * 8]);
      const uint4 b1 = *(const uint4*)(&Bs[cur][col][16 + half * 8]);
      uintx8 w;
      w[0] = b0.x; w[1] = b0.y; w[2] = b0.z; w[3] = b0.w;
      w[4] = b1.x; w[5] = b1.y; w[6] = b1.z; w[7] = b1.w;
      bfr[tj] = __builtin_bit_cast(bf16x16, w);
    }
#pragma unroll
    for (int ti = 0; ti < 2; ++ti)
#pragma unroll
      for (int tj = 0; tj < 4; ++tj)
        acc[ti][tj] = __builtin_amdgcn_wmma_f32_16x16x32_bf16(
            false, af[ti], false, bfr[tj], (short)0, acc[ti][tj], false, false);

    asm volatile("s_wait_asynccnt 0x0" ::: "memory");
    __syncthreads();   // next buffer ready for all; this buffer free to overwrite
  }

  // ---- store C: element i of lane -> row = 8*half + i, col = l16 ----
#pragma unroll
  for (int ti = 0; ti < 2; ++ti)
#pragma unroll
    for (int tj = 0; tj < 4; ++tj)
#pragma unroll
      for (int i = 0; i < 8; ++i) {
        int row = bm + wm * 32 + ti * 16 + half * 8 + i;
        int col = bn + wn * 64 + tj * 16 + l16;
        if (mfull || row < M) {
          float v = acc[ti][tj][i];
          if (bias) v += bias[col];
          if (act_gelu) v = gelu_f(v);
          if (Cf) Cf[(size_t)row * N + col] = v;
          if (Cb) Cb[(size_t)row * N + col] = f2bf(v);
        }
      }
}

// ---------------- naive fp32 GEMM (tiny N projections, heads) ----------------
__global__ void gemm_naive_f32_k(const float* __restrict__ A, const float* __restrict__ W,
                                 const float* __restrict__ bias, float* __restrict__ C,
                                 int M, int N, int K, int act_gelu)
{
  int idx = blockIdx.x * blockDim.x + threadIdx.x;
  if (idx >= M * N) return;
  int m = idx / N, n = idx % N;
  const float* a = A + (size_t)m * K;
  const float* w = W + (size_t)n * K;
  float s = bias ? bias[n] : 0.f;
  for (int k = 0; k < K; ++k) s += a[k] * w[k];
  if (act_gelu) s = gelu_f(s);
  C[idx] = s;
}

// ---------------- elementwise helpers ----------------
__global__ void f32_to_bf16_k(const float* __restrict__ s, unsigned short* __restrict__ d, size_t n) {
  size_t i = (size_t)blockIdx.x * blockDim.x + threadIdx.x;
  if (i < n) d[i] = f2bf(s[i]);
}
__global__ void add_inplace_k(float* __restrict__ y, const float* __restrict__ x, int n) {
  int i = blockIdx.x * blockDim.x + threadIdx.x;
  if (i < n) y[i] += x[i];
}
__global__ void add_vec_k(const float* __restrict__ a, const float* __restrict__ b,
                          float* __restrict__ o, int n) {
  int i = blockIdx.x * blockDim.x + threadIdx.x;
  if (i < n) o[i] = a[i] + b[i];
}
__global__ void q_init_k(const float* __restrict__ query, float* __restrict__ q) {
  int i = blockIdx.x * blockDim.x + threadIdx.x;
  if (i >= ROWS_ * D_) return;
  int row = i >> 10, d = i & (D_ - 1);
  q[i] = query[(row & 1) * D_ + d];
}
__global__ void center_combine_k(const float* __restrict__ c01, const float* __restrict__ delta,
                                 float* __restrict__ center) {
  int i = threadIdx.x;
  if (i < 2 * B_) {
    float v = c01[i] + delta[i];
    center[i] = fminf(fmaxf(v, EPSC_), 1.f - EPSC_);
  }
}

// ---------------- patch extraction + normalize -> bf16 matrix ----------------
__global__ __launch_bounds__(256) void prep_patches_k(const float* __restrict__ x,
                                                      unsigned short* __restrict__ out) {
  int m = blockIdx.x;                                   // token row
  int b = m / NTOK_, n = m % NTOK_;
  int gy = n / HP_, gx = n % HP_;
  const float mean[3] = {0.485f, 0.456f, 0.406f};
  const float isd[3]  = {1.f / 0.229f, 1.f / 0.224f, 1.f / 0.225f};
  unsigned short* o = out + (size_t)m * (3 * P_ * P_);
  for (int k = threadIdx.x; k < 3 * P_ * P_; k += 256) {
    int c = k >> 8, rem = k & 255, py = rem >> 4, px = rem & 15;
    float v = x[(((size_t)b * 3 + c) * IMG_ + (gy * P_ + py)) * IMG_ + (gx * P_ + px)];
    o[k] = f2bf((v - mean[c]) * isd[c]);
  }
}

// ---------------- center estimation ----------------
__global__ __launch_bounds__(256) void center_logits_k(const float* __restrict__ tokens,
                                                       const float* __restrict__ cmw,
                                                       const float* __restrict__ cmb,
                                                       float* __restrict__ logits) {
  int tok = blockIdx.x * 8 + (threadIdx.x >> 5);
  int lane = threadIdx.x & 31;
  if (tok >= MTOK_) return;
  const float* t = tokens + (size_t)tok * CB_;
  float s = 0.f;
  for (int c = lane; c < CB_; c += 32) s += t[c] * cmw[c];
  for (int m = 16; m; m >>= 1) s += __shfl_xor(s, m, 32);
  if (lane == 0) logits[tok] = s + cmb[0];
}

__global__ __launch_bounds__(256) void soft_argmax_k(const float* __restrict__ lg,
                                                     float* __restrict__ c01) {
  __shared__ float sh[256];
  int b = blockIdx.x, tid = threadIdx.x;
  const float* l = lg + b * NTOK_;
  float m = -3.0e38f;
  for (int i = tid; i < NTOK_; i += 256) m = fmaxf(m, l[i]);
  sh[tid] = m; __syncthreads();
  for (int s = 128; s > 0; s >>= 1) { if (tid < s) sh[tid] = fmaxf(sh[tid], sh[tid + s]); __syncthreads(); }
  float mx = sh[0]; __syncthreads();
  float se = 0.f, sx = 0.f, sy = 0.f;
  for (int i = tid; i < NTOK_; i += 256) {
    float e = expf(l[i] - mx);
    int h = i / HP_, w = i % HP_;
    se += e; sx += e * ((w + 0.5f) * (1.f / HP_)); sy += e * ((h + 0.5f) * (1.f / HP_));
  }
  sh[tid] = se; __syncthreads();
  for (int s = 128; s > 0; s >>= 1) { if (tid < s) sh[tid] += sh[tid + s]; __syncthreads(); }
  float SE = sh[0]; __syncthreads();
  sh[tid] = sx; __syncthreads();
  for (int s = 128; s > 0; s >>= 1) { if (tid < s) sh[tid] += sh[tid + s]; __syncthreads(); }
  float SX = sh[0]; __syncthreads();
  sh[tid] = sy; __syncthreads();
  for (int s = 128; s > 0; s >>= 1) { if (tid < s) sh[tid] += sh[tid + s]; __syncthreads(); }
  float SY = sh[0];
  if (tid == 0) { c01[b * 2] = SX / SE; c01[b * 2 + 1] = SY / SE; }
}

__global__ void token_mean_k(const float* __restrict__ tokens, float* __restrict__ h0) {
  int idx = blockIdx.x * blockDim.x + threadIdx.x;
  if (idx >= B_ * CB_) return;
  int b = idx / CB_, c = idx % CB_;
  const float* p = tokens + (size_t)b * NTOK_ * CB_ + c;
  float s = 0.f;
  for (int n = 0; n < NTOK_; ++n) s += p[(size_t)n * CB_];
  h0[idx] = s * (1.f / NTOK_);
}

// ---------------- row LayerNorm (outputs f32 and/or bf16) ----------------
__global__ __launch_bounds__(256) void layernorm_rows_k(const float* __restrict__ X,
    const float* __restrict__ g, const float* __restrict__ bta,
    float* __restrict__ Yf, unsigned short* __restrict__ Yb, int W)
{
  __shared__ float sh[256];
  int row = blockIdx.x, tid = threadIdx.x;
  const float* x = X + (size_t)row * W;
  float s = 0.f;
  for (int i = tid; i < W; i += 256) s += x[i];
  sh[tid] = s; __syncthreads();
  for (int st = 128; st > 0; st >>= 1) { if (tid < st) sh[tid] += sh[tid + st]; __syncthreads(); }
  float mu = sh[0] / W; __syncthreads();
  float v = 0.f;
  for (int i = tid; i < W; i += 256) { float d = x[i] - mu; v += d * d; }
  sh[tid] = v; __syncthreads();
  for (int st = 128; st > 0; st >>= 1) { if (tid < st) sh[tid] += sh[tid + st]; __syncthreads(); }
  float rstd = rsqrtf(sh[0] / W + 1e-5f);
  for (int i = tid; i < W; i += 256) {
    float y = (x[i] - mu) * rstd * g[i] + bta[i];
    if (Yf) Yf[(size_t)row * W + i] = y;
    if (Yb) Yb[(size_t)row * W + i] = f2bf(y);
  }
}

// ---------------- polar resample: cart (B,N,D) -> polar bf16 (B,T,R,D) ----------------
__global__ __launch_bounds__(256) void polar_sample_k(const float* __restrict__ tok,
                                                      const float* __restrict__ center,
                                                      unsigned short* __restrict__ polar) {
  int pr = blockIdx.x;                         // (b*T + t)*R + r
  int b = pr >> 11, t = (pr >> 5) & (T_ - 1), r = pr & (R_ - 1);
  float cx = center[b * 2], cy = center[b * 2 + 1];
  float th = (float)t * (6.283185307179586f / T_);
  float rad = (float)r * (1.f / (R_ - 1));
  float px = cx + rad * cosf(th);
  float py = cy + rad * sinf(th);
  float gx = px * (float)HP_ - 0.5f, gy = py * (float)HP_ - 0.5f;   // (g+1)*W/2-0.5, g=p*2-1
  float x0f = floorf(gx), y0f = floorf(gy);
  float wx = gx - x0f, wy = gy - y0f;
  int x0 = clampi((int)x0f, 0, HP_ - 1), x1 = clampi((int)x0f + 1, 0, HP_ - 1);
  int y0 = clampi((int)y0f, 0, HP_ - 1), y1 = clampi((int)y0f + 1, 0, HP_ - 1);
  const float* base = tok + (size_t)b * NTOK_ * D_;
  size_t i00 = (size_t)(y0 * HP_ + x0) * D_, i01 = (size_t)(y0 * HP_ + x1) * D_;
  size_t i10 = (size_t)(y1 * HP_ + x0) * D_, i11 = (size_t)(y1 * HP_ + x1) * D_;
  float w00 = (1 - wx) * (1 - wy), w01 = wx * (1 - wy), w10 = (1 - wx) * wy, w11 = wx * wy;
  unsigned short* o = polar + (size_t)pr * D_;
  for (int d = threadIdx.x; d < D_; d += 256)
    o[d] = f2bf(w00 * base[i00 + d] + w01 * base[i01 + d] +
                w10 * base[i10 + d] + w11 * base[i11 + d]);
}

// ---------------- depthwise 3x3 (theta wrap, r edge) + GELU ----------------
__global__ void dwconv_gelu_k(const unsigned short* __restrict__ polar,
                              const float* __restrict__ dww, const float* __restrict__ dwb,
                              unsigned short* __restrict__ mixed) {
  size_t idx = (size_t)blockIdx.x * blockDim.x + threadIdx.x;
  if (idx >= (size_t)MPOL_ * D_) return;
  int d = (int)(idx & (D_ - 1));
  size_t pr = idx >> 10;
  int r = (int)(pr & (R_ - 1)), t = (int)((pr >> 5) & (T_ - 1)), b = (int)(pr >> 11);
  float s = dwb[d];
#pragma unroll
  for (int i = 0; i < 3; ++i) {
    int tt = t + i - 1; tt = (tt + T_) & (T_ - 1);
#pragma unroll
    for (int j = 0; j < 3; ++j) {
      int rr = clampi(r + j - 1, 0, R_ - 1);
      s += dww[d * 9 + i * 3 + j] *
           bf2f(polar[(((size_t)(b * T_ + tt) * R_) + rr) * D_ + d]);
    }
  }
  mixed[idx] = f2bf(gelu_f(s));
}

// ---------------- multi-head attention on 2 queries ----------------
__global__ __launch_bounds__(128) void mha_k(const float* __restrict__ qkv, float* __restrict__ o) {
  int b = blockIdx.x >> 3, h = blockIdx.x & 7;
  int d = threadIdx.x;                       // 0..127
  __shared__ float sq[2][DH_], sk[2][DH_], sv[2][DH_];
  __shared__ float sc[2][2], aw[2][2];
  for (int qi = 0; qi < 2; ++qi) {
    const float* base = qkv + (size_t)(b * 2 + qi) * (3 * D_) + h * DH_;
    sq[qi][d] = base[d]; sk[qi][d] = base[D_ + d]; sv[qi][d] = base[2 * D_ + d];
  }
  __syncthreads();
  if (d < 4) {
    int qi = d >> 1, ki = d & 1;
    float s = 0.f;
    for (int k = 0; k < DH_; ++k) s += sq[qi][k] * sk[ki][k];
    sc[qi][ki] = s * 0.08838834764831845f;   // 1/sqrt(128)
  }
  __syncthreads();
  if (d < 2) {
    float m = fmaxf(sc[d][0], sc[d][1]);
    float e0 = expf(sc[d][0] - m), e1 = expf(sc[d][1] - m);
    float inv = 1.f / (e0 + e1);
    aw[d][0] = e0 * inv; aw[d][1] = e1 * inv;
  }
  __syncthreads();
  for (int qi = 0; qi < 2; ++qi)
    o[(size_t)(b * 2 + qi) * D_ + h * DH_ + d] = aw[qi][0] * sv[0][d] + aw[qi][1] * sv[1][d];
}

// ---------------- deformable cross-attention sampling ----------------
__global__ __launch_bounds__(256) void deform_attn_k(
    const float* __restrict__ rd, const float* __restrict__ off, const float* __restrict__ alog,
    const float* __restrict__ center, const float* __restrict__ tok,
    const unsigned short* __restrict__ polar2, const float* __restrict__ level_embed,
    float* __restrict__ out)
{
  int row = blockIdx.x;                       // b*QN + q
  int b = row >> 1;
  __shared__ float att[16], lgs[16];
  __shared__ size_t c00[16], c01[16], c10[16], c11[16];
  __shared__ float w00s[16], w01s[16], w10s[16], w11s[16];
  __shared__ int lvlS[16];
  int tid = threadIdx.x;
  if (tid < 16) {
    int lvl = tid >> 3, p = tid & 7;
    lgs[tid] = alog[row * 16 + tid];
    float bx = (lvl == 0) ? center[b * 2] : 0.5f;
    float by = (lvl == 0) ? center[b * 2 + 1] : 0.5f;
    bx = fminf(fmaxf(bx, EPSC_), 1.f - EPSC_);
    by = fminf(fmaxf(by, EPSC_), 1.f - EPSC_);
    float refx = 1.f / (1.f + expf(-(logf(bx / (1.f - bx)) + rd[row * 4 + lvl * 2])));
    float refy = 1.f / (1.f + expf(-(logf(by / (1.f - by)) + rd[row * 4 + lvl * 2 + 1])));
    int Wl = (lvl == 0) ? HP_ : R_;
    int Hl = (lvl == 0) ? HP_ : T_;
    float lx = refx + off[row * 32 + (lvl * NPTS_ + p) * 2] / (float)Wl;
    float ly = refy + off[row * 32 + (lvl * NPTS_ + p) * 2 + 1] / (float)Hl;
    if (lvl == 1) ly = ly - floorf(ly);       // wrap theta
    float gx = lx * (float)Wl - 0.5f, gy = ly * (float)Hl - 0.5f;
    float x0f = floorf(gx), y0f = floorf(gy);
    float wx = gx - x0f, wy = gy - y0f;
    int x0 = clampi((int)x0f, 0, Wl - 1), x1 = clampi((int)x0f + 1, 0, Wl - 1);
    int y0 = clampi((int)y0f, 0, Hl - 1), y1 = clampi((int)y0f + 1, 0, Hl - 1);
    size_t rb = (lvl == 0) ? (size_t)b * NTOK_ : (size_t)b * T_ * R_;
    int Wrow = (lvl == 0) ? HP_ : R_;
    c00[tid] = (rb + (size_t)y0 * Wrow + x0) * D_;
    c01[tid] = (rb + (size_t)y0 * Wrow + x1) * D_;
    c10[tid] = (rb + (size_t)y1 * Wrow + x0) * D_;
    c11[tid] = (rb + (size_t)y1 * Wrow + x1) * D_;
    w00s[tid] = (1 - wx) * (1 - wy); w01s[tid] = wx * (1 - wy);
    w10s[tid] = (1 - wx) * wy;       w11s[tid] = wx * wy;
    lvlS[tid] = lvl;
  }
  __syncthreads();
  if (tid == 0) {                             // joint softmax over all L*NPTS logits
    float m = -3.0e38f;
    for (int s = 0; s < 16; ++s) m = fmaxf(m, lgs[s]);
    float se = 0.f;
    for (int s = 0; s < 16; ++s) { att[s] = expf(lgs[s] - m); se += att[s]; }
    float inv = 1.f / se;
    for (int s = 0; s < 16; ++s) att[s] *= inv;
  }
  __syncthreads();
  for (int d = tid; d < D_; d += 256) {
    float acc = 0.f;
    float le0 = level_embed[d];               // maps[0] = cart + level_embed[0]
#pragma unroll
    for (int s = 0; s < 16; ++s) {
      float v;
      if (lvlS[s] == 0) {
        v = w00s[s] * tok[c00[s] + d] + w01s[s] * tok[c01[s] + d] +
            w10s[s] * tok[c10[s] + d] + w11s[s] * tok[c11[s] + d] + le0;
      } else {                                // level_embed[1] folded into pw bias
        v = w00s[s] * bf2f(polar2[c00[s] + d]) + w01s[s] * bf2f(polar2[c01[s] + d]) +
            w10s[s] * bf2f(polar2[c10[s] + d]) + w11s[s] * bf2f(polar2[c11[s] + d]);
      }
      acc += att[s] * v;
    }
    out[(size_t)row * D_ + d] = acc;
  }
}

// ---------------- host orchestration ----------------
extern "C" void kernel_launch(void* const* d_in, const int* in_sizes, int n_in,
                              void* d_out, int out_size, void* d_ws, size_t ws_size,
                              hipStream_t stream) {
  (void)in_sizes; (void)n_in; (void)out_size; (void)ws_size;
  auto F = [&](int i) { return (const float*)d_in[i]; };
  // param indices (setup_inputs dict order, blocks flattened)
  enum { IX_X = 0, IX_PE_W, IX_PE_B, IX_INP_W, IX_INP_B, IX_CH_LN_G, IX_CH_LN_B,
         IX_CH_W1, IX_CH_B1, IX_CH_W2, IX_CH_B2, IX_CM_W, IX_CM_B, IX_DW_W, IX_DW_B,
         IX_PW_W, IX_PW_B, IX_LE, IX_QUERY, IX_BLK0 };
  const int BSTR = 22;
  const int IX_OUT_LN_G = IX_BLK0 + 2 * BSTR;   // 63

  // ---- workspace carve-up (aliased live ranges) ----
  const size_t SZ_PATCH = (size_t)MTOK_ * CB_ * 2;
  const size_t SZ_TOKF  = (size_t)MTOK_ * CB_ * 4;
  const size_t SZ_R1    = SZ_PATCH + SZ_TOKF + SZ_PATCH;     // 77MB, also hosts polar/polar2
  const size_t SZ_MIXED = (size_t)MPOL_ * D_ * 2;
  const size_t SZ_TOKD  = (size_t)MTOK_ * D_ * 4;
  char* W = (char*)d_ws;
  size_t off = 0;
  auto take = [&](size_t bytes) -> char* {
    char* p = W + off; off += (bytes + 255) & ~(size_t)255; return p;
  };
  char* r1 = take(SZ_R1);
  unsigned short* patchA   = (unsigned short*)r1;
  float*          tokensF  = (float*)(r1 + SZ_PATCH);
  unsigned short* tokensUS = (unsigned short*)(r1 + SZ_PATCH + SZ_TOKF);
  unsigned short* polarUS  = (unsigned short*)r1;     // after tokens are dead
  unsigned short* polar2US = (unsigned short*)r1;     // after polar is dead
  unsigned short* mixedUS  = (unsigned short*)take(SZ_MIXED);
  float*          tokD     = (float*)take(SZ_TOKD);   // cart map (B,N,D), live to the end
  unsigned short* peUS  = (unsigned short*)take((size_t)CB_ * 3 * P_ * P_ * 2);
  unsigned short* inpUS = (unsigned short*)take((size_t)D_ * CB_ * 2);
  unsigned short* pwUS  = (unsigned short*)take((size_t)D_ * D_ * 2);
  unsigned short* decUS[2][5];
  const size_t decN[5] = {(size_t)3 * D_ * D_, (size_t)D_ * D_, (size_t)D_ * D_,
                          (size_t)DFF_ * D_, (size_t)D_ * DFF_};
  for (int bp = 0; bp < 2; ++bp)
    for (int j = 0; j < 5; ++j) decUS[bp][j] = (unsigned short*)take(decN[j] * 2);
  float* logits = (float*)take((size_t)MTOK_ * 4);
  float* c01v   = (float*)take(256);
  float* centerv= (float*)take(256);
  float* h0     = (float*)take((size_t)B_ * CB_ * 4);
  float* hn     = (float*)take((size_t)B_ * CB_ * 4);
  float* h1c    = (float*)take((size_t)B_ * (CB_ / 2) * 4);
  float* deltav = (float*)take(256);
  float* pwb2   = (float*)take((size_t)D_ * 4);
  float* qbuf   = (float*)take((size_t)ROWS_ * D_ * 4);
  float* qnf    = (float*)take((size_t)ROWS_ * D_ * 4);
  unsigned short* qnus = (unsigned short*)take((size_t)ROWS_ * D_ * 2);
  float* qkv    = (float*)take((size_t)ROWS_ * 3 * D_ * 4);
  float* mo     = (float*)take((size_t)ROWS_ * D_ * 4);
  unsigned short* mous = (unsigned short*)take((size_t)ROWS_ * D_ * 2);
  float* tmpf   = (float*)take((size_t)ROWS_ * D_ * 4);
  float* rdb    = (float*)take((size_t)ROWS_ * 4 * 4);
  float* offb   = (float*)take((size_t)ROWS_ * 32 * 4);
  float* ab     = (float*)take((size_t)ROWS_ * 16 * 4);
  float* dof    = (float*)take((size_t)ROWS_ * D_ * 4);
  unsigned short* dous = (unsigned short*)take((size_t)ROWS_ * D_ * 2);
  unsigned short* ffhus = (unsigned short*)take((size_t)ROWS_ * DFF_ * 2);
  float* lnf    = (float*)take((size_t)ROWS_ * D_ * 4);
  float* h1o    = (float*)take((size_t)ROWS_ * (D_ / 2) * 4);

  auto conv = [&](const float* s, unsigned short* d, size_t n) {
    f32_to_bf16_k<<<dim3((unsigned)((n + 255) / 256)), 256, 0, stream>>>(s, d, n);
  };
  auto gemm = [&](const unsigned short* A, const unsigned short* Bw, const float* bias,
                  float* Cf, unsigned short* Cb, int M, int N, int K, int gelu) {
    dim3 g((N + GN - 1) / GN, (M + GM - 1) / GM);
    gemm_bf16_wmma_k<<<g, 256, 0, stream>>>(A, Bw, bias, Cf, Cb, M, N, K, gelu);
  };
  auto ngemm = [&](const float* A, const float* Wt, const float* bias, float* C,
                   int M, int N, int K, int gelu) {
    gemm_naive_f32_k<<<(M * N + 255) / 256, 256, 0, stream>>>(A, Wt, bias, C, M, N, K, gelu);
  };

  // 0) weight conversions to bf16
  conv(F(IX_PE_W), peUS, (size_t)CB_ * 3 * P_ * P_);
  conv(F(IX_INP_W), inpUS, (size_t)D_ * CB_);
  conv(F(IX_PW_W), pwUS, (size_t)D_ * D_);
  for (int bp = 0; bp < 2; ++bp) {
    int base = IX_BLK0 + bp * BSTR;
    conv(F(base + 6), decUS[bp][0], decN[0]);
    conv(F(base + 8), decUS[bp][1], decN[1]);
    conv(F(base + 16), decUS[bp][2], decN[2]);
    conv(F(base + 18), decUS[bp][3], decN[3]);
    conv(F(base + 20), decUS[bp][4], decN[4]);
  }

  // 1) patch matrix + patch-embed GEMM -> tokens (f32 + bf16)
  prep_patches_k<<<MTOK_, 256, 0, stream>>>(F(IX_X), patchA);
  gemm(patchA, peUS, F(IX_PE_B), tokensF, tokensUS, MTOK_, CB_, 3 * P_ * P_, 0);

  // 2) center estimation
  center_logits_k<<<MTOK_ / 8, 256, 0, stream>>>(tokensF, F(IX_CM_W), F(IX_CM_B), logits);
  soft_argmax_k<<<B_, 256, 0, stream>>>(logits, c01v);
  token_mean_k<<<(B_ * CB_ + 255) / 256, 256, 0, stream>>>(tokensF, h0);
  layernorm_rows_k<<<B_, 256, 0, stream>>>(h0, F(IX_CH_LN_G), F(IX_CH_LN_B), hn, nullptr, CB_);
  ngemm(hn, F(IX_CH_W1), F(IX_CH_B1), h1c, B_, CB_ / 2, CB_, 1);
  ngemm(h1c, F(IX_CH_W2), F(IX_CH_B2), deltav, B_, 2, CB_ / 2, 0);
  center_combine_k<<<1, 64, 0, stream>>>(c01v, deltav, centerv);

  // 3) input projection -> cart map (B,N,D)
  gemm(tokensUS, inpUS, F(IX_INP_B), tokD, nullptr, MTOK_, D_, CB_, 0);

  // 4) polar resample -> dwconv+gelu -> pointwise WMMA GEMM (bias + level_embed[1])
  polar_sample_k<<<MPOL_, 256, 0, stream>>>(tokD, centerv, polarUS);
  dwconv_gelu_k<<<(unsigned)(((size_t)MPOL_ * D_ + 255) / 256), 256, 0, stream>>>(
      polarUS, F(IX_DW_W), F(IX_DW_B), mixedUS);
  add_vec_k<<<4, 256, 0, stream>>>(F(IX_PW_B), F(IX_LE) + D_, pwb2, D_);
  gemm(mixedUS, pwUS, pwb2, nullptr, polar2US, MPOL_, D_, D_, 0);

  // 5) decoder
  q_init_k<<<(ROWS_ * D_ + 255) / 256, 256, 0, stream>>>(F(IX_QUERY), qbuf);
  for (int bp = 0; bp < 2; ++bp) {
    int base = IX_BLK0 + bp * BSTR;
    layernorm_rows_k<<<ROWS_, 256, 0, stream>>>(qbuf, F(base + 0), F(base + 1), nullptr, qnus, D_);
    gemm(qnus, decUS[bp][0], F(base + 7), qkv, nullptr, ROWS_, 3 * D_, D_, 0);
    mha_k<<<B_ * NH_, 128, 0, stream>>>(qkv, mo);
    conv(mo, mous, (size_t)ROWS_ * D_);
    gemm(mous, decUS[bp][1], F(base + 9), tmpf, nullptr, ROWS_, D_, D_, 0);
    add_inplace_k<<<(ROWS_ * D_) / 256, 256, 0, stream>>>(qbuf, tmpf, ROWS_ * D_);

    layernorm_rows_k<<<ROWS_, 256, 0, stream>>>(qbuf, F(base + 2), F(base + 3), qnf, nullptr, D_);
    ngemm(qnf, F(base + 10), F(base + 11), rdb, ROWS_, 4, D_, 0);
    ngemm(qnf, F(base + 12), F(base + 13), offb, ROWS_, 32, D_, 0);
    ngemm(qnf, F(base + 14), F(base + 15), ab, ROWS_, 16, D_, 0);
    deform_attn_k<<<ROWS_, 256, 0, stream>>>(rdb, offb, ab, centerv, tokD, polar2US,
                                             F(IX_LE), dof);
    conv(dof, dous, (size_t)ROWS_ * D_);
    gemm(dous, decUS[bp][2], F(base + 17), tmpf, nullptr, ROWS_, D_, D_, 0);
    add_inplace_k<<<(ROWS_ * D_) / 256, 256, 0, stream>>>(qbuf, tmpf, ROWS_ * D_);

    layernorm_rows_k<<<ROWS_, 256, 0, stream>>>(qbuf, F(base + 4), F(base + 5), nullptr, qnus, D_);
    gemm(qnus, decUS[bp][3], F(base + 19), nullptr, ffhus, ROWS_, DFF_, D_, 1);
    gemm(ffhus, decUS[bp][4], F(base + 21), tmpf, nullptr, ROWS_, D_, DFF_, 0);
    add_inplace_k<<<(ROWS_ * D_) / 256, 256, 0, stream>>>(qbuf, tmpf, ROWS_ * D_);
  }

  // 6) output head -> (B,4)
  layernorm_rows_k<<<ROWS_, 256, 0, stream>>>(qbuf, F(IX_OUT_LN_G), F(IX_OUT_LN_G + 1),
                                              lnf, nullptr, D_);
  ngemm(lnf, F(IX_OUT_LN_G + 2), F(IX_OUT_LN_G + 3), h1o, ROWS_, D_ / 2, D_, 1);
  ngemm(h1o, F(IX_OUT_LN_G + 4), F(IX_OUT_LN_G + 5), (float*)d_out, ROWS_, 2, D_ / 2, 0);
}
